// SpatialContextAggregation_47691316855218
// MI455X (gfx1250) — matrix operand order repaired and checked by reading, not compiled, not run on hardware
//
#include <hip/hip_runtime.h>
#include <hip/hip_bf16.h>
#include <math.h>

// ---------------------------------------------------------------------------
// SpatialContextAggregation for MI455X (gfx1250, wave32, WMMA).
// Pipeline: FPS -> gather -> curvature(PCA ball) -> cost -> sinkhorn(20) ->
//           gamma^T@feats (WMMA) -> sim (WMMA) -> topk-agg ->
//           feats@p^T (WMMA) -> sinkhorn(20) -> w@p (WMMA) + residual.
// lp matrix: 65537 x 516 (interior 65536 x 512, pad col @512, pad row @65536,
// cols 513..515 always zero, stride chosen for 16B-aligned b128 access).
// ---------------------------------------------------------------------------

#define N_PTS   65536
#define M_PROTO 512
#define C_FEAT  128
#define LDM     516            // lp row stride (16B-aligned rows)
#define LP_ROWS 65537          // N_PTS + 1 pad row
#define NB      10
#define TOPK    20
#define LDT     36             // LDS tile stride (f16): 72B rows -> 8B-aligned h4

static constexpr float TAUv   = 0.05f;
static constexpr float SCv    = 0.08838834764831845f;   // 1/sqrt(128)
static constexpr float ISQ3   = 0.5773502691896258f;    // 1/sqrt(3)
static constexpr float EPSv   = 1e-8f;

typedef __attribute__((ext_vector_type(16))) _Float16 v16h;
typedef __attribute__((ext_vector_type(8)))  float    v8f;
typedef __attribute__((ext_vector_type(4)))  _Float16 h4;

// ---- WMMA 16x16x32 f16 step. A: LDS row-major 16x32 (lda elems/row).
// ---- B: LDS column-major (Bt[n][k], ldb elems/column). Layouts per ISA 7.12.2.
__device__ __forceinline__ void wmma_step(const _Float16* __restrict__ A, int lda,
                                          const _Float16* __restrict__ B, int ldb,
                                          v8f& acc, int lane) {
  const int m = lane & 15;       // A row / B column
  const int h = lane >> 4;       // lane half selects K sub-range
  v16h a, b;
#pragma unroll
  for (int j = 0; j < 16; ++j) {
    const int v = j >> 1;
    const int k = ((v < 4) ? 0 : 16) + h * 8 + ((v & 3) << 1) + (j & 1);
    a[j] = A[m * lda + k];                 // A[m][k]
    b[j] = B[m * ldb + k];                 // B^T[n][k]
  }
  acc = __builtin_amdgcn_wmma_f32_16x16x32_f16(false, a, false, b,
                                               (short)0, acc, false, false);
}

// ---------------------------------------------------------------------------
// 1) Farthest point sampling. 1 block, 1024 threads, 512 iterations.
// ---------------------------------------------------------------------------
__global__ __launch_bounds__(1024)
void fps_kernel(const float* __restrict__ xyz, float* __restrict__ dist,
                int* __restrict__ proto) {
  __shared__ float sval[1024];
  __shared__ int   sidx[1024];
  __shared__ float cent[3];
  __shared__ int   farsh;
  const int tid = threadIdx.x;
  for (int p = tid; p < N_PTS; p += 1024) dist[p] = 1e10f;
  if (tid == 0) farsh = 0;
  __syncthreads();
  for (int it = 0; it < M_PROTO; ++it) {
    if (tid == 0) {
      const int f = farsh;
      proto[it] = f;
      cent[0] = xyz[f * 3 + 0]; cent[1] = xyz[f * 3 + 1]; cent[2] = xyz[f * 3 + 2];
    }
    __syncthreads();
    const float cx = cent[0], cy = cent[1], cz = cent[2];
    float bv = -1.0f; int bi = 0;
    for (int j = 0; j < 64; ++j) {
      const int p = j * 1024 + tid;
      const float dx = xyz[p * 3 + 0] - cx;
      const float dy = xyz[p * 3 + 1] - cy;
      const float dz = xyz[p * 3 + 2] - cz;
      const float d2 = dx * dx + dy * dy + dz * dz;
      const float dd = fminf(dist[p], d2);
      dist[p] = dd;
      if (dd > bv) { bv = dd; bi = p; }
    }
    sval[tid] = bv; sidx[tid] = bi;
    __syncthreads();
    for (int s = 512; s > 0; s >>= 1) {
      if (tid < s) {
        const float ov = sval[tid + s]; const int oi = sidx[tid + s];
        if (ov > sval[tid] || (ov == sval[tid] && oi < sidx[tid])) {
          sval[tid] = ov; sidx[tid] = oi;
        }
      }
      __syncthreads();
    }
    if (tid == 0) farsh = sidx[0];
    __syncthreads();
  }
}

// ---------------------------------------------------------------------------
// 2) Gather prototypes + mean of prototype xyz. 1 block, 512 threads.
// ---------------------------------------------------------------------------
__global__ __launch_bounds__(512)
void gather_kernel(const float* __restrict__ xyz, const float* __restrict__ feats,
                   const int* __restrict__ proto, float* __restrict__ pxyz,
                   float* __restrict__ pfa, float* __restrict__ meanw) {
  __shared__ float red[512];
  const int tid = threadIdx.x;
  const int f = proto[tid];
  const float px = xyz[f * 3 + 0], py = xyz[f * 3 + 1], pz = xyz[f * 3 + 2];
  pxyz[tid * 3 + 0] = px; pxyz[tid * 3 + 1] = py; pxyz[tid * 3 + 2] = pz;
  for (int c = 0; c < C_FEAT; c += 4) {
    const float4 v = *(const float4*)(feats + (size_t)f * C_FEAT + c);
    *(float4*)(pfa + (size_t)tid * C_FEAT + c) = v;
  }
  float comp[3] = {px, py, pz};
  for (int d = 0; d < 3; ++d) {
    red[tid] = comp[d];
    __syncthreads();
    for (int s = 256; s > 0; s >>= 1) {
      if (tid < s) red[tid] += red[tid + s];
      __syncthreads();
    }
    if (tid == 0) meanw[d] = red[0] / 512.0f;
    __syncthreads();
  }
}

// ---------------------------------------------------------------------------
// 3) Curvature via PCA on ball neighborhoods. One block per prototype.
// ---------------------------------------------------------------------------
__device__ __forceinline__ void eig3_sym(float a00, float a01, float a02,
                                         float a11, float a12, float a22,
                                         float& l0, float& l1, float& l2) {
  const float p1 = a01 * a01 + a02 * a02 + a12 * a12;
  const float q  = (a00 + a11 + a22) / 3.0f;
  const float b00 = a00 - q, b11 = a11 - q, b22 = a22 - q;
  const float p2 = b00 * b00 + b11 * b11 + b22 * b22 + 2.0f * p1;
  const float p  = sqrtf(fmaxf(p2, 0.0f) / 6.0f);
  if (p < 1e-20f) {
    float x = a00, y = a11, z = a22, t;
    if (x > y) { t = x; x = y; y = t; }
    if (y > z) { t = y; y = z; z = t; }
    if (x > y) { t = x; x = y; y = t; }
    l0 = x; l1 = y; l2 = z;
    return;
  }
  const float ip = 1.0f / p;
  const float c00 = b00 * ip, c01 = a01 * ip, c02 = a02 * ip;
  const float c11 = b11 * ip, c12 = a12 * ip, c22 = b22 * ip;
  float r = 0.5f * (c00 * (c11 * c22 - c12 * c12)
                  - c01 * (c01 * c22 - c12 * c02)
                  + c02 * (c01 * c12 - c11 * c02));
  r = fminf(1.0f, fmaxf(-1.0f, r));
  const float phi = acosf(r) / 3.0f;
  const float e1 = q + 2.0f * p * cosf(phi);
  const float e3 = q + 2.0f * p * cosf(phi + 2.0943951023931953f);
  const float e2 = 3.0f * q - e1 - e3;
  l0 = e3; l1 = e2; l2 = e1;
}

__global__ __launch_bounds__(256)
void curv_kernel(const float* __restrict__ xyz, const float* __restrict__ pxyz,
                 const float* __restrict__ meanw, float* __restrict__ distge) {
  __shared__ int   cand[256 * NB];
  __shared__ int   cnt[256];
  __shared__ int   off[256];
  __shared__ int   list[NB];
  __shared__ int   totalSh;
  const int tid = threadIdx.x;
  const int m   = blockIdx.x;
  const float qx = pxyz[m * 3 + 0], qy = pxyz[m * 3 + 1], qz = pxyz[m * 3 + 2];
  int c = 0;
  const int base = tid * 256;
  for (int j = 0; j < 256; ++j) {
    const int i = base + j;
    const float dx = qx - xyz[i * 3 + 0];
    const float dy = qy - xyz[i * 3 + 1];
    const float dz = qz - xyz[i * 3 + 2];
    const float d = sqrtf(fmaxf(dx * dx + dy * dy + dz * dz, 0.0f));
    if (d <= 0.04f && c < NB) { cand[tid * NB + c] = i; ++c; }
  }
  cnt[tid] = c;
  __syncthreads();
  if (tid == 0) {
    int o = 0;
    for (int t = 0; t < 256; ++t) { off[t] = o; o += cnt[t]; }
    totalSh = o;
  }
  __syncthreads();
  for (int j = 0; j < c; ++j) {
    const int pos = off[tid] + j;
    if (pos < NB) list[pos] = cand[tid * NB + j];
  }
  __syncthreads();
  if (tid == 0) {
    int total = totalSh; if (total > NB) total = NB;
    const int fill = (total > 0) ? list[0] : N_PTS;
    for (int j = total; j < NB; ++j) list[j] = fill;
    const float mx = meanw[0], my = meanw[1], mz = meanw[2];
    float s00 = 0, s01 = 0, s02 = 0, s11 = 0, s12 = 0, s22 = 0;
    for (int j = 0; j < NB; ++j) {
      const int li = list[j];
      float nx, ny, nz;
      if (li < N_PTS) { nx = xyz[li * 3 + 0]; ny = xyz[li * 3 + 1]; nz = xyz[li * 3 + 2]; }
      else            { nx = mx; ny = my; nz = mz; }
      const float cx = nx - qx, cy = ny - qy, cz = nz - qz;
      s00 += cx * cx; s01 += cx * cy; s02 += cx * cz;
      s11 += cy * cy; s12 += cy * cz; s22 += cz * cz;
    }
    const float inb = 1.0f / (float)NB;
    s00 = s00 * inb + EPSv; s01 = s01 * inb + EPSv; s02 = s02 * inb + EPSv;
    s11 = s11 * inb + EPSv; s12 = s12 * inb + EPSv; s22 = s22 * inb + EPSv;
    float l0, l1, l2;
    eig3_sym(s00, s01, s02, s11, s12, s22, l0, l1, l2);
    const float l2c = fmaxf(l2, EPSv);
    const float f0 = (l2c - l1) / l2c;
    const float f1 = (l1 - l0) / l2c;
    const float f2 = l0 / l2c;
    distge[m] = sqrtf(f0 * f0 + f1 * f1 + f2 * f2) * ISQ3;
  }
}

// ---------------------------------------------------------------------------
// 4) lp[n][m] = -(||xyz_n - p_m|| * sc + dist_ge[m]) / TAU ; all pads = 0.
// ---------------------------------------------------------------------------
__global__ __launch_bounds__(256)
void cost_kernel(const float* __restrict__ xyz, const float* __restrict__ pxyz,
                 const float* __restrict__ distge, float* __restrict__ lp) {
  __shared__ float sp[M_PROTO * 3];
  __shared__ float sg[M_PROTO];
  const int tid = threadIdx.x;
  for (int e = tid; e < M_PROTO * 3; e += 256) sp[e] = pxyz[e];
  for (int e = tid; e < M_PROTO;     e += 256) sg[e] = distge[e];
  __syncthreads();
  const size_t idx = (size_t)blockIdx.x * 256 + tid;
  if (idx >= (size_t)LP_ROWS * LDM) return;
  const int n = (int)(idx / LDM), m = (int)(idx % LDM);
  float v = 0.0f;
  if (n < N_PTS && m < M_PROTO) {
    const float dx = xyz[n * 3 + 0] - sp[m * 3 + 0];
    const float dy = xyz[n * 3 + 1] - sp[m * 3 + 1];
    const float dz = xyz[n * 3 + 2] - sp[m * 3 + 2];
    const float d = sqrtf(fmaxf(dx * dx + dy * dy + dz * dz, 0.0f));
    v = -(d * SCv + sg[m]) / TAUv;
  }
  lp[idx] = v;
}

// ---------------------------------------------------------------------------
// 5) Sinkhorn. Row: one wave per row, 16 contiguous cols per lane (b128 IO),
//    lane 0 owns the pad column. Col: one block per 4 columns, float4 sweeps.
// ---------------------------------------------------------------------------
__global__ __launch_bounds__(256)
void sk_row_kernel(float* __restrict__ lp) {
  const int lane = threadIdx.x & 31;
  const int row = blockIdx.x * 8 + (threadIdx.x >> 5);   // rows 0..N-1 only
  float* base = lp + (size_t)row * LDM;
  float4 v[4];
  float mx = -1e30f;
#pragma unroll
  for (int i = 0; i < 4; ++i) {
    v[i] = *(const float4*)(base + lane * 16 + i * 4);
    mx = fmaxf(mx, fmaxf(fmaxf(v[i].x, v[i].y), fmaxf(v[i].z, v[i].w)));
  }
  const float vex = (lane == 0) ? base[M_PROTO] : -1e30f;
  mx = fmaxf(mx, vex);
  for (int o = 16; o > 0; o >>= 1) mx = fmaxf(mx, __shfl_xor(mx, o, 32));
  float s = 0.0f;
#pragma unroll
  for (int i = 0; i < 4; ++i)
    s += __expf(v[i].x - mx) + __expf(v[i].y - mx) + __expf(v[i].z - mx) + __expf(v[i].w - mx);
  if (lane == 0) s += __expf(vex - mx);
  for (int o = 16; o > 0; o >>= 1) s += __shfl_xor(s, o, 32);
  const float lse = mx + __logf(s);
#pragma unroll
  for (int i = 0; i < 4; ++i) {
    v[i].x -= lse; v[i].y -= lse; v[i].z -= lse; v[i].w -= lse;
    *(float4*)(base + lane * 16 + i * 4) = v[i];
  }
  if (lane == 0) base[M_PROTO] = vex - lse;
}

__global__ __launch_bounds__(256)
void sk_col_kernel(float* __restrict__ lp) {
  __shared__ float4 red[256];
  const int tid = threadIdx.x;
  const int c0 = blockIdx.x * 4;                 // cols 0..M-1 only
  float4 mx = make_float4(-1e30f, -1e30f, -1e30f, -1e30f);
  for (int r = tid; r < LP_ROWS; r += 256) {
    const float4 v = *(const float4*)(lp + (size_t)r * LDM + c0);
    mx.x = fmaxf(mx.x, v.x); mx.y = fmaxf(mx.y, v.y);
    mx.z = fmaxf(mx.z, v.z); mx.w = fmaxf(mx.w, v.w);
  }
  red[tid] = mx; __syncthreads();
  for (int s = 128; s > 0; s >>= 1) {
    if (tid < s) {
      float4 a = red[tid], b = red[tid + s];
      a.x = fmaxf(a.x, b.x); a.y = fmaxf(a.y, b.y);
      a.z = fmaxf(a.z, b.z); a.w = fmaxf(a.w, b.w);
      red[tid] = a;
    }
    __syncthreads();
  }
  mx = red[0]; __syncthreads();
  float4 sum = make_float4(0.f, 0.f, 0.f, 0.f);
  for (int r = tid; r < LP_ROWS; r += 256) {
    const float4 v = *(const float4*)(lp + (size_t)r * LDM + c0);
    sum.x += __expf(v.x - mx.x); sum.y += __expf(v.y - mx.y);
    sum.z += __expf(v.z - mx.z); sum.w += __expf(v.w - mx.w);
  }
  red[tid] = sum; __syncthreads();
  for (int s = 128; s > 0; s >>= 1) {
    if (tid < s) {
      float4 a = red[tid], b = red[tid + s];
      a.x += b.x; a.y += b.y; a.z += b.z; a.w += b.w;
      red[tid] = a;
    }
    __syncthreads();
  }
  const float4 ss = red[0];
  const float4 lse = make_float4(mx.x + __logf(ss.x), mx.y + __logf(ss.y),
                                 mx.z + __logf(ss.z), mx.w + __logf(ss.w));
  for (int r = tid; r < LP_ROWS; r += 256) {
    float4 v = *(const float4*)(lp + (size_t)r * LDM + c0);
    v.x -= lse.x; v.y -= lse.y; v.z -= lse.z; v.w -= lse.w;
    *(float4*)(lp + (size_t)r * LDM + c0) = v;
  }
}

__global__ __launch_bounds__(256)
void pad_kernel(float* __restrict__ lp) {
  const int i = blockIdx.x * 256 + threadIdx.x;
  if (i < LP_ROWS) lp[(size_t)i * LDM + M_PROTO] = 0.0f;
  if (i < LDM)     lp[(size_t)N_PTS * LDM + i]   = 0.0f;
}

// ---------------------------------------------------------------------------
// 6) WMMA GEMMs. Each block: 256 thr = 8 waves, 64x64 output, 2 tiles/wave.
// ---------------------------------------------------------------------------
// pfb[m][c] = 0.5*pfa[m][c] + 0.5 * sum_n exp(lp[n][m]) * feats[n][c]
__global__ __launch_bounds__(256)
void agg1_gemm(const float* __restrict__ lp, const float* __restrict__ feats,
               const float* __restrict__ pfa, float* __restrict__ pfb) {
  __shared__ _Float16 As[64 * LDT];
  __shared__ _Float16 Bs[64 * LDT];
  const int tid = threadIdx.x, lane = tid & 31, wave = tid >> 5;
  const int m0 = blockIdx.x * 64, c0 = blockIdx.y * 64;
  v8f acc0 = {}, acc1 = {};
  for (int k0 = 0; k0 < N_PTS; k0 += 32) {
    __syncthreads();
    for (int e = tid; e < 512; e += 256) {          // b128 staging, 2 iters
      const int kk = e >> 4, q = e & 15;            // m = q*4 .. q*4+3
      const float4 ga = *(const float4*)(lp + (size_t)(k0 + kk) * LDM + m0 + q * 4);
      As[(q * 4 + 0) * LDT + kk] = (_Float16)__expf(ga.x);
      As[(q * 4 + 1) * LDT + kk] = (_Float16)__expf(ga.y);
      As[(q * 4 + 2) * LDT + kk] = (_Float16)__expf(ga.z);
      As[(q * 4 + 3) * LDT + kk] = (_Float16)__expf(ga.w);
      const float4 fb = *(const float4*)(feats + (size_t)(k0 + kk) * C_FEAT + c0 + q * 4);
      Bs[(q * 4 + 0) * LDT + kk] = (_Float16)fb.x;
      Bs[(q * 4 + 1) * LDT + kk] = (_Float16)fb.y;
      Bs[(q * 4 + 2) * LDT + kk] = (_Float16)fb.z;
      Bs[(q * 4 + 3) * LDT + kk] = (_Float16)fb.w;
      if (k0 + 32 < N_PTS)                          // prefetch next K chunk
        __builtin_prefetch((const void*)(lp + (size_t)(k0 + 32 + kk) * LDM + m0 + q * 4), 0, 1);
    }
    __syncthreads();
    int t = wave, tr = t >> 2, tc = t & 3;
    wmma_step(As + tr * 16 * LDT, LDT, Bs + tc * 16 * LDT, LDT, acc0, lane);
    t = wave + 8; tr = t >> 2; tc = t & 3;
    wmma_step(As + tr * 16 * LDT, LDT, Bs + tc * 16 * LDT, LDT, acc1, lane);
  }
  const int h = lane >> 4, nn = lane & 15;
  int t = wave, tr = t >> 2, tc = t & 3;
#pragma unroll
  for (int r = 0; r < 8; ++r) {
    const int gm = m0 + tr * 16 + h * 8 + r, gc = c0 + tc * 16 + nn;
    pfb[(size_t)gm * C_FEAT + gc] = 0.5f * pfa[(size_t)gm * C_FEAT + gc] + 0.5f * acc0[r];
  }
  t = wave + 8; tr = t >> 2; tc = t & 3;
#pragma unroll
  for (int r = 0; r < 8; ++r) {
    const int gm = m0 + tr * 16 + h * 8 + r, gc = c0 + tc * 16 + nn;
    pfb[(size_t)gm * C_FEAT + gc] = 0.5f * pfa[(size_t)gm * C_FEAT + gc] + 0.5f * acc1[r];
  }
}

// sim[m][k] = sc * <pfb[m], pfb[k]>
__global__ __launch_bounds__(256)
void sim_gemm(const float* __restrict__ pfb, float* __restrict__ sim) {
  __shared__ _Float16 As[64 * LDT];
  __shared__ _Float16 Bs[64 * LDT];
  const int tid = threadIdx.x, lane = tid & 31, wave = tid >> 5;
  const int m0 = blockIdx.x * 64, n0 = blockIdx.y * 64;
  v8f acc0 = {}, acc1 = {};
  for (int k0 = 0; k0 < C_FEAT; k0 += 32) {
    __syncthreads();
    for (int e = tid; e < 512; e += 256) {
      const int rr = e >> 3, q = e & 7;             // k = q*4 .. q*4+3 (contig)
      const float4 a = *(const float4*)(pfb + (size_t)(m0 + rr) * C_FEAT + k0 + q * 4);
      h4 ha; ha.x = (_Float16)a.x; ha.y = (_Float16)a.y; ha.z = (_Float16)a.z; ha.w = (_Float16)a.w;
      *(h4*)(As + rr * LDT + q * 4) = ha;
      const float4 b = *(const float4*)(pfb + (size_t)(n0 + rr) * C_FEAT + k0 + q * 4);
      h4 hb; hb.x = (_Float16)b.x; hb.y = (_Float16)b.y; hb.z = (_Float16)b.z; hb.w = (_Float16)b.w;
      *(h4*)(Bs + rr * LDT + q * 4) = hb;
    }
    __syncthreads();
    int t = wave, tr = t >> 2, tc = t & 3;
    wmma_step(As + tr * 16 * LDT, LDT, Bs + tc * 16 * LDT, LDT, acc0, lane);
    t = wave + 8; tr = t >> 2; tc = t & 3;
    wmma_step(As + tr * 16 * LDT, LDT, Bs + tc * 16 * LDT, LDT, acc1, lane);
  }
  const int h = lane >> 4, nn = lane & 15;
  int t = wave, tr = t >> 2, tc = t & 3;
#pragma unroll
  for (int r = 0; r < 8; ++r)
    sim[(size_t)(m0 + tr * 16 + h * 8 + r) * M_PROTO + (n0 + tc * 16 + nn)] = SCv * acc0[r];
  t = wave + 8; tr = t >> 2; tc = t & 3;
#pragma unroll
  for (int r = 0; r < 8; ++r)
    sim[(size_t)(m0 + tr * 16 + h * 8 + r) * M_PROTO + (n0 + tc * 16 + nn)] = SCv * acc1[r];
}

// top-20 of sim[:,k] (ties -> lowest index), weight-normalized aggregation.
__global__ __launch_bounds__(128)
void topk_kernel(const float* __restrict__ sim, const float* __restrict__ pfb,
                 float* __restrict__ pfc) {
  __shared__ float sv[M_PROTO];
  __shared__ float rv[128];
  __shared__ int   ri[128];
  __shared__ float tv[TOPK];
  __shared__ int   ti[TOPK];
  const int tid = threadIdx.x;
  const int k = blockIdx.x;
  for (int m = tid; m < M_PROTO; m += 128) sv[m] = sim[(size_t)m * M_PROTO + k];
  __syncthreads();
  for (int j = 0; j < TOPK; ++j) {
    float bv = -1e30f; int bi = 1 << 30;
    for (int i = 0; i < 4; ++i) {
      const int m = tid + i * 128;
      const float v = sv[m];
      if (v > bv || (v == bv && m < bi)) { bv = v; bi = m; }
    }
    rv[tid] = bv; ri[tid] = bi;
    __syncthreads();
    for (int s = 64; s > 0; s >>= 1) {
      if (tid < s) {
        const float ov = rv[tid + s]; const int oi = ri[tid + s];
        if (ov > rv[tid] || (ov == rv[tid] && oi < ri[tid])) { rv[tid] = ov; ri[tid] = oi; }
      }
      __syncthreads();
    }
    if (tid == 0) { tv[j] = rv[0]; ti[j] = ri[0]; sv[ri[0]] = -1e30f; }
    __syncthreads();
  }
  float S = 0.0f;
  for (int j = 0; j < TOPK; ++j) S += tv[j];
  const float Sc = fmaxf(S, 1e-4f);
  const int c = tid;
  float agg = 0.0f;
  for (int j = 0; j < TOPK; ++j) agg += (tv[j] / Sc) * pfb[(size_t)ti[j] * C_FEAT + c];
  pfc[(size_t)k * C_FEAT + c] = 0.5f * pfb[(size_t)k * C_FEAT + c] + 0.5f * agg;
}

// lp[n][m] = -(sc/TAU) * <feats[n], pfc[m]>   (interior only)
__global__ __launch_bounds__(256)
void simpt_gemm(const float* __restrict__ feats, const float* __restrict__ pfc,
                float* __restrict__ lp) {
  __shared__ _Float16 As[64 * LDT];
  __shared__ _Float16 Bs[64 * LDT];
  const int tid = threadIdx.x, lane = tid & 31, wave = tid >> 5;
  const int n0 = blockIdx.x * 64, m0 = blockIdx.y * 64;
  v8f acc0 = {}, acc1 = {};
  for (int k0 = 0; k0 < C_FEAT; k0 += 32) {
    __syncthreads();
    for (int e = tid; e < 512; e += 256) {
      const int rr = e >> 3, q = e & 7;
      const float4 a = *(const float4*)(feats + (size_t)(n0 + rr) * C_FEAT + k0 + q * 4);
      h4 ha; ha.x = (_Float16)a.x; ha.y = (_Float16)a.y; ha.z = (_Float16)a.z; ha.w = (_Float16)a.w;
      *(h4*)(As + rr * LDT + q * 4) = ha;
      const float4 b = *(const float4*)(pfc + (size_t)(m0 + rr) * C_FEAT + k0 + q * 4);
      h4 hb; hb.x = (_Float16)b.x; hb.y = (_Float16)b.y; hb.z = (_Float16)b.z; hb.w = (_Float16)b.w;
      *(h4*)(Bs + rr * LDT + q * 4) = hb;
    }
    __syncthreads();
    int t = wave, tr = t >> 2, tc = t & 3;
    wmma_step(As + tr * 16 * LDT, LDT, Bs + tc * 16 * LDT, LDT, acc0, lane);
    t = wave + 8; tr = t >> 2; tc = t & 3;
    wmma_step(As + tr * 16 * LDT, LDT, Bs + tc * 16 * LDT, LDT, acc1, lane);
  }
  const float scale = -SCv / TAUv;
  const int h = lane >> 4, nn = lane & 15;
  int t = wave, tr = t >> 2, tc = t & 3;
#pragma unroll
  for (int r = 0; r < 8; ++r)
    lp[(size_t)(n0 + tr * 16 + h * 8 + r) * LDM + (m0 + tc * 16 + nn)] = scale * acc0[r];
  t = wave + 8; tr = t >> 2; tc = t & 3;
#pragma unroll
  for (int r = 0; r < 8; ++r)
    lp[(size_t)(n0 + tr * 16 + h * 8 + r) * LDM + (m0 + tc * 16 + nn)] = scale * acc1[r];
}

// out[n][c] = 0.5*feats[n][c] + 0.5 * sum_m exp(lp[n][m]) * pfc[m][c]
__global__ __launch_bounds__(256)
void refine_gemm(const float* __restrict__ lp, const float* __restrict__ pfc,
                 const float* __restrict__ feats, float* __restrict__ out) {
  __shared__ _Float16 As[64 * LDT];
  __shared__ _Float16 Bs[64 * LDT];
  const int tid = threadIdx.x, lane = tid & 31, wave = tid >> 5;
  const int n0 = blockIdx.x * 64, c0 = blockIdx.y * 64;
  v8f acc0 = {}, acc1 = {};
  for (int k0 = 0; k0 < M_PROTO; k0 += 32) {
    __syncthreads();
    for (int e = tid; e < 512; e += 256) {
      const int rr = e >> 3, q = e & 7;             // A: k contiguous
      const float4 a = *(const float4*)(lp + (size_t)(n0 + rr) * LDM + k0 + q * 4);
      h4 ha;
      ha.x = (_Float16)__expf(a.x); ha.y = (_Float16)__expf(a.y);
      ha.z = (_Float16)__expf(a.z); ha.w = (_Float16)__expf(a.w);
      *(h4*)(As + rr * LDT + q * 4) = ha;
      const int kk = e >> 4, qc = e & 15;           // B: c contiguous
      const float4 b = *(const float4*)(pfc + (size_t)(k0 + kk) * C_FEAT + c0 + qc * 4);
      Bs[(qc * 4 + 0) * LDT + kk] = (_Float16)b.x;
      Bs[(qc * 4 + 1) * LDT + kk] = (_Float16)b.y;
      Bs[(qc * 4 + 2) * LDT + kk] = (_Float16)b.z;
      Bs[(qc * 4 + 3) * LDT + kk] = (_Float16)b.w;
    }
    __syncthreads();
    int t = wave, tr = t >> 2, tc = t & 3;
    wmma_step(As + tr * 16 * LDT, LDT, Bs + tc * 16 * LDT, LDT, acc0, lane);
    t = wave + 8; tr = t >> 2; tc = t & 3;
    wmma_step(As + tr * 16 * LDT, LDT, Bs + tc * 16 * LDT, LDT, acc1, lane);
  }
  const int h = lane >> 4, nn = lane & 15;
  int t = wave, tr = t >> 2, tc = t & 3;
#pragma unroll
  for (int r = 0; r < 8; ++r) {
    const int gn = n0 + tr * 16 + h * 8 + r, gc = c0 + tc * 16 + nn;
    out[(size_t)gn * C_FEAT + gc] = 0.5f * feats[(size_t)gn * C_FEAT + gc] + 0.5f * acc0[r];
  }
  t = wave + 8; tr = t >> 2; tc = t & 3;
#pragma unroll
  for (int r = 0; r < 8; ++r) {
    const int gn = n0 + tr * 16 + h * 8 + r, gc = c0 + tc * 16 + nn;
    out[(size_t)gn * C_FEAT + gc] = 0.5f * feats[(size_t)gn * C_FEAT + gc] + 0.5f * acc1[r];
  }
}

// ---------------------------------------------------------------------------
extern "C" void kernel_launch(void* const* d_in, const int* in_sizes, int n_in,
                              void* d_out, int out_size, void* d_ws, size_t ws_size,
                              hipStream_t stream) {
  const float* xyz   = (const float*)d_in[0];   // 65536 x 3
  const float* feats = (const float*)d_in[1];   // 65536 x 128
  float* out = (float*)d_out;                   // 65536 x 128

  // workspace layout (floats)
  float* lp     = (float*)d_ws;                          // 65537*516
  float* dist   = lp + (size_t)LP_ROWS * LDM;            // 65536
  int*   proto  = (int*)(dist + N_PTS);                  // 512
  float* pxyz   = (float*)(proto + M_PROTO);             // 512*3
  float* pfa    = pxyz + M_PROTO * 3;                    // 512*128
  float* pfb    = pfa + M_PROTO * C_FEAT;                // 512*128
  float* pfc    = pfb + M_PROTO * C_FEAT;                // 512*128
  float* meanw  = pfc + M_PROTO * C_FEAT;                // 4
  float* distge = meanw + 4;                             // 512
  float* simbuf = distge + M_PROTO;                      // 512*512

  fps_kernel<<<1, 1024, 0, stream>>>(xyz, dist, proto);
  gather_kernel<<<1, 512, 0, stream>>>(xyz, feats, proto, pxyz, pfa, meanw);
  curv_kernel<<<M_PROTO, 256, 0, stream>>>(xyz, pxyz, meanw, distge);

  {
    const size_t total = (size_t)LP_ROWS * LDM;
    const int blocks = (int)((total + 255) / 256);
    cost_kernel<<<blocks, 256, 0, stream>>>(xyz, pxyz, distge, lp);
  }
  for (int i = 0; i < 20; ++i) {
    sk_row_kernel<<<N_PTS / 8, 256, 0, stream>>>(lp);
    sk_col_kernel<<<M_PROTO / 4, 256, 0, stream>>>(lp);
  }

  agg1_gemm<<<dim3(M_PROTO / 64, C_FEAT / 64), 256, 0, stream>>>(lp, feats, pfa, pfb);
  sim_gemm<<<dim3(M_PROTO / 64, M_PROTO / 64), 256, 0, stream>>>(pfb, simbuf);
  topk_kernel<<<M_PROTO, 128, 0, stream>>>(simbuf, pfb, pfc);

  simpt_gemm<<<dim3(N_PTS / 64, M_PROTO / 64), 256, 0, stream>>>(feats, pfc, lp);
  pad_kernel<<<(LP_ROWS + 255) / 256, 256, 0, stream>>>(lp);
  for (int i = 0; i < 20; ++i) {
    sk_row_kernel<<<N_PTS / 8, 256, 0, stream>>>(lp);
    sk_col_kernel<<<M_PROTO / 4, 256, 0, stream>>>(lp);
  }

  refine_gemm<<<dim3(N_PTS / 64, C_FEAT / 64), 256, 0, stream>>>(lp, pfc, feats, out);
}
